// ExtremeOptimizationLayer_50964081934697
// MI455X (gfx1250) — matrix operand order repaired and checked by reading, not compiled, not run on hardware
//
#include <hip/hip_runtime.h>

// ---------------------------------------------------------------------------
// MI455X (gfx1250): bf16 WMMA (16x16x32, f32 accum) for both large GEMMs
// (~550 GFLOP vs ~1GB HBM traffic -> compute bound; bf16 = 8x FLOP/instr vs
// fp32 WMMA 16x16x4). BatchNorm folded into Wf1 scaling + bias. Weights are
// pre-transposed once so GEMM2/GEMM3 tiles are verbatim 16B copies, enabling
// GLOBAL_LOAD_ASYNC_TO_LDS_B128 double-buffered staging (ASYNCcnt-tracked).
// ---------------------------------------------------------------------------

#define K_BRANCH 64
#define D_IN     1024
#define D_OUT    1024
#define BATCH    2048
#define KO       (K_BRANCH * D_OUT)   // 65536
#define SPLITK   8
#define EPSV     1e-5f

typedef __attribute__((ext_vector_type(16))) __bf16 v16bf;
typedef __attribute__((ext_vector_type(8)))  float  v8f;
typedef __attribute__((ext_vector_type(4)))  int    v4i;

#if defined(__gfx1250__) && __has_builtin(__builtin_amdgcn_global_load_async_to_lds_b128)
#define USE_ASYNC 1
#else
#define USE_ASYNC 0
#endif

// fp32 -> bf16, round-to-nearest-even
__device__ __forceinline__ unsigned short f32_to_bf16(float f) {
  union { float f; unsigned u; } v; v.f = f;
  unsigned r = v.u + 0x7FFFu + ((v.u >> 16) & 1u);
  return (unsigned short)(r >> 16);
}
__device__ __forceinline__ float bf16_to_f32(unsigned short u) {
  union { unsigned u; float f; } v; v.u = ((unsigned)u) << 16;
  return v.f;
}

// 16-byte tile copy global->LDS: async (ASYNCcnt path) or sync fallback.
__device__ __forceinline__ void copy16(const unsigned short* g, unsigned short* l) {
#if USE_ASYNC
  typedef __attribute__((address_space(1))) v4i gv4i;
  typedef __attribute__((address_space(3))) v4i lv4i;
  __builtin_amdgcn_global_load_async_to_lds_b128(
      (gv4i*)(unsigned long long)g,
      (lv4i*)(unsigned)(unsigned long long)l,
      0, 0);
#else
  *(uint4*)l = *(const uint4*)g;
#endif
}
__device__ __forceinline__ void async_wait(int n) {
#if USE_ASYNC
#if __has_builtin(__builtin_amdgcn_s_wait_asynccnt)
  if (n == 0) __builtin_amdgcn_s_wait_asynccnt(0);
  else        __builtin_amdgcn_s_wait_asynccnt(4);
#else
  if (n == 0) asm volatile("s_wait_asynccnt 0x0" ::: "memory");
  else        asm volatile("s_wait_asynccnt 0x4" ::: "memory");
#endif
#endif
}

// ---- WMMA fragment loads from LDS (ISA 7.12.2 bf16 layouts) ----------------
// A (16x32 MxK) and B (32x16 KxN stored as [n][k]): both LDS pitch-32 bf16.
__device__ __forceinline__ v16bf load_frag_a(const unsigned short* base,
                                             int laneIdx, int laneHalf) {
  const unsigned short* row = base + laneIdx * 32;
  union { uint4 u[2]; v16bf v; } f;
  f.u[0] = *(const uint4*)(row + (laneHalf ? 8  : 0));
  f.u[1] = *(const uint4*)(row + (laneHalf ? 24 : 16));
  return f.v;
}
__device__ __forceinline__ v16bf load_frag_b(const unsigned short* base,
                                             int laneIdx, int laneHalf) {
  const unsigned short* row = base + laneIdx * 32;
  union { uint4 u[2]; v16bf v; } f;
  f.u[0] = *(const uint4*)(row + (laneHalf ? 8  : 0));
  f.u[1] = *(const uint4*)(row + (laneHalf ? 24 : 16));
  return f.v;
}

// 8-accumulator micro-kernel: wave tile 32(M) x 64(N), K=32 step.
__device__ __forceinline__ void wmma_tile8(const unsigned short* ldsA,
                                           const unsigned short* ldsB,
                                           int waveRow, int waveCol,
                                           int laneIdx, int laneHalf,
                                           v8f acc[2][4]) {
  v16bf a[2], b[4];
#pragma unroll
  for (int i = 0; i < 2; ++i)
    a[i] = load_frag_a(ldsA + (waveRow * 32 + i * 16) * 32, laneIdx, laneHalf);
#pragma unroll
  for (int j = 0; j < 4; ++j)
    b[j] = load_frag_b(ldsB + (waveCol * 64 + j * 16) * 32, laneIdx, laneHalf);
#pragma unroll
  for (int i = 0; i < 2; ++i)
#pragma unroll
    for (int j = 0; j < 4; ++j)
      acc[i][j] = __builtin_amdgcn_wmma_f32_16x16x32_bf16(
          false, a[i], false, b[j], (short)0, acc[i][j], false, false);
}

// ---------------------------------------------------------------------------
// Kernel 1: h[b, k*O+o] = bf16( relu( x[b,:] . W[k][:,o] + bias[k,o] ) )
// f32 sources need conversion -> synchronous staging.
// ---------------------------------------------------------------------------
__global__ __launch_bounds__(256) void gemm1_kernel(
    const float* __restrict__ x, const float* __restrict__ W,
    const float* __restrict__ bb, unsigned short* __restrict__ h) {
  __shared__ unsigned short ldsA[128 * 32];
  __shared__ unsigned short ldsB[128 * 32];
  const int c0 = blockIdx.x * 128;
  const int r0 = blockIdx.y * 128;
  const int kb = blockIdx.z;
  const int tid = threadIdx.x;
  const int lane = tid & 31, wave = tid >> 5;
  const int waveRow = wave & 3, waveCol = wave >> 2;
  const int laneIdx = lane & 15, laneHalf = lane >> 4;
  const float* Wk = W + (size_t)kb * D_IN * D_OUT;

  v8f acc[2][4] = {};

  for (int kt = 0; kt < D_IN / 32; ++kt) {
    const int kk = kt * 32;
    // stage A: x tile [128 x 32] f32 -> bf16 LDS row-major (b64 stores)
#pragma unroll
    for (int i = 0; i < 4; ++i) {
      int idx = tid + i * 256;        // 0..1023
      int row = idx >> 3, kq = idx & 7;
      const float* gp = x + (size_t)(r0 + row) * D_IN + kk + kq * 4;
      if (kt + 1 < D_IN / 32) __builtin_prefetch(gp + 32, 0, 0);
      float4 f = *(const float4*)gp;
      uint2 p;
      p.x = (unsigned)f32_to_bf16(f.x) | ((unsigned)f32_to_bf16(f.y) << 16);
      p.y = (unsigned)f32_to_bf16(f.z) | ((unsigned)f32_to_bf16(f.w) << 16);
      *(uint2*)&ldsA[row * 32 + kq * 4] = p;
    }
    // stage B: W_k [32 x 128] -> LDS [n][k]; 4x4 block per thread,
    // 4 coalesced float4 loads + 4 b64 LDS stores (k-contiguous).
    {
      int nq = tid & 31;              // n-quad: cols nq*4..nq*4+3
      int kq = tid >> 5;              // k-quad: rows kq*4..kq*4+3
      float4 fr[4];
#pragma unroll
      for (int r = 0; r < 4; ++r) {
        const float* gp = Wk + (size_t)(kk + kq * 4 + r) * D_OUT + c0 + nq * 4;
        if (kt + 1 < D_IN / 32) __builtin_prefetch(gp + 32 * D_OUT, 0, 0);
        fr[r] = *(const float4*)gp;
      }
#pragma unroll
      for (int n = 0; n < 4; ++n) {
        float e0 = (&fr[0].x)[n], e1 = (&fr[1].x)[n];
        float e2 = (&fr[2].x)[n], e3 = (&fr[3].x)[n];
        uint2 p;
        p.x = (unsigned)f32_to_bf16(e0) | ((unsigned)f32_to_bf16(e1) << 16);
        p.y = (unsigned)f32_to_bf16(e2) | ((unsigned)f32_to_bf16(e3) << 16);
        *(uint2*)&ldsB[(nq * 4 + n) * 32 + kq * 4] = p;
      }
    }
    __syncthreads();
    wmma_tile8(ldsA, ldsB, waveRow, waveCol, laneIdx, laneHalf, acc);
    __syncthreads();
  }

  // epilogue: bias + ReLU -> bf16 h
#pragma unroll
  for (int j = 0; j < 4; ++j) {
    int colo = c0 + waveCol * 64 + j * 16 + laneIdx;
    float bias_v = bb[kb * D_OUT + colo];
    size_t colIdx = (size_t)kb * D_OUT + colo;
#pragma unroll
    for (int i = 0; i < 2; ++i) {
      int rowBase = r0 + waveRow * 32 + i * 16 + laneHalf * 8;
#pragma unroll
      for (int r = 0; r < 8; ++r) {
        float v = fmaxf(acc[i][j][r] + bias_v, 0.f);
        h[(size_t)(rowBase + r) * KO + colIdx] = f32_to_bf16(v);
      }
    }
  }
}

// ---------------------------------------------------------------------------
// Kernel 2: per-column (k,o) batch stats -> scale/shift (deterministic).
// ---------------------------------------------------------------------------
__global__ __launch_bounds__(256) void stats_kernel(
    const unsigned short* __restrict__ h, const float* __restrict__ gamma,
    const float* __restrict__ beta, float* __restrict__ scale,
    float* __restrict__ shift) {
  __shared__ float redS[128], redQ[128];
  const int t = threadIdx.x;
  const int col = blockIdx.x * 128 + (t & 127);
  const int half = t >> 7;
  float s = 0.f, q = 0.f;
  for (int row = half; row < BATCH; row += 2) {
    float v = bf16_to_f32(h[(size_t)row * KO + col]);
    s += v; q += v * v;
  }
  if (half) { redS[t & 127] = s; redQ[t & 127] = q; }
  __syncthreads();
  if (!half) {
    s += redS[t]; q += redQ[t];
    float mean = s * (1.f / BATCH);
    float var  = fmaxf(q * (1.f / BATCH) - mean * mean, 0.f);
    float sc   = gamma[col] * rsqrtf(var + EPSV);
    scale[col] = sc;
    shift[col] = beta[col] - mean * sc;
  }
}

// ---------------------------------------------------------------------------
// Kernel 3: wf1sT[j, c] = bf16(scale[c]*Wf1[c, j])  (pre-TRANSPOSED so GEMM2's
// B staging is a verbatim copy); partB[cb, j] = sum_c shift[c]*Wf1[c, j].
// ---------------------------------------------------------------------------
__global__ __launch_bounds__(256) void wf1prep_kernel(
    const float* __restrict__ Wf1, const float* __restrict__ scale,
    const float* __restrict__ shift, unsigned short* __restrict__ wf1sT,
    float* __restrict__ partB) {
  const int j  = blockIdx.x * 256 + threadIdx.x;   // 0..1023
  const int cb = blockIdx.y;                        // 0..63
  float acc = 0.f;
  union { unsigned short s[8]; uint4 u; } pk;
  for (int ci = 0; ci < KO / 64; ci += 8) {
#pragma unroll
    for (int u = 0; u < 8; ++u) {
      int c = cb * (KO / 64) + ci + u;
      float w = Wf1[(size_t)c * D_OUT + j];
      pk.s[u] = f32_to_bf16(scale[c] * w);
      acc += shift[c] * w;
    }
    *(uint4*)&wf1sT[(size_t)j * KO + cb * (KO / 64) + ci] = pk.u;
  }
  partB[cb * D_OUT + j] = acc;
}

__global__ __launch_bounds__(256) void bias2_reduce_kernel(
    const float* __restrict__ partB, const float* __restrict__ bf1,
    float* __restrict__ bias2) {
  const int j = blockIdx.x * 256 + threadIdx.x;
  float acc = bf1[j];
  for (int cb = 0; cb < 64; ++cb) acc += partB[cb * D_OUT + j];
  bias2[j] = acc;
}

// Wf2 [k][n] -> wf2sT[n][k] bf16 (transposed convert; reads coalesced)
__global__ __launch_bounds__(256) void wf2prep_kernel(
    const float* __restrict__ Wf2, unsigned short* __restrict__ wf2sT) {
  int idx = blockIdx.x * 256 + threadIdx.x;
  int k = idx >> 10, n = idx & (D_OUT - 1);
  wf2sT[(size_t)n * D_OUT + k] = f32_to_bf16(Wf2[idx]);
}

// ---------------------------------------------------------------------------
// Kernel 4: split-K GEMM2 partials: zpart[s] = h_bf16 . Wf1sT^T
// Double-buffered LDS; async global->LDS staging (4 x b128 per thread/stage).
// ---------------------------------------------------------------------------
__global__ __launch_bounds__(256) void gemm2_kernel(
    const unsigned short* __restrict__ hA, const unsigned short* __restrict__ wBT,
    float* __restrict__ zpart) {
  __shared__ unsigned short ldsA[2][128 * 32];
  __shared__ unsigned short ldsB[2][128 * 32];
  const int c0 = blockIdx.x * 128;
  const int r0 = blockIdx.y * 128;
  const int kbase = blockIdx.z * (KO / SPLITK);
  const int tid = threadIdx.x;
  const int lane = tid & 31, wave = tid >> 5;
  const int waveRow = wave & 3, waveCol = wave >> 2;
  const int laneIdx = lane & 15, laneHalf = lane >> 4;
  const int iters = (KO / SPLITK) / 32;

  v8f acc[2][4] = {};

  auto stage = [&](int buf, int kk) {
#pragma unroll
    for (int i = 0; i < 2; ++i) {
      int idx = tid + i * 256;          // 0..511
      int row = idx >> 2, kc = idx & 3;
      copy16(hA + (size_t)(r0 + row) * KO + kk + kc * 8,
             &ldsA[buf][row * 32 + kc * 8]);
    }
#pragma unroll
    for (int i = 0; i < 2; ++i) {
      int idx = tid + i * 256;          // 0..511 (row = n index)
      int row = idx >> 2, kc = idx & 3;
      copy16(wBT + (size_t)(c0 + row) * KO + kk + kc * 8,
             &ldsB[buf][row * 32 + kc * 8]);
    }
  };

  stage(0, kbase);
  for (int kt = 0; kt < iters; ++kt) {
    int buf = kt & 1;
    if (kt + 1 < iters) {
      stage(buf ^ 1, kbase + (kt + 1) * 32);
      async_wait(4);          // retire previous stage's 4 copies (in-order)
    } else {
      async_wait(0);
    }
    __syncthreads();
    wmma_tile8(ldsA[buf], ldsB[buf], waveRow, waveCol, laneIdx, laneHalf, acc);
    __syncthreads();
  }

  float* zp = zpart + (size_t)blockIdx.z * BATCH * D_OUT;
#pragma unroll
  for (int j = 0; j < 4; ++j) {
    int colo = c0 + waveCol * 64 + j * 16 + laneIdx;
#pragma unroll
    for (int i = 0; i < 2; ++i) {
      int rowBase = r0 + waveRow * 32 + i * 16 + laneHalf * 8;
#pragma unroll
      for (int r = 0; r < 8; ++r)
        zp[(size_t)(rowBase + r) * D_OUT + colo] = acc[i][j][r];
    }
  }
}

// z = relu(sum_s zpart[s] + bias2) -> bf16
__global__ __launch_bounds__(256) void zreduce_kernel(
    const float* __restrict__ zpart, const float* __restrict__ bias2,
    unsigned short* __restrict__ zbf) {
  const int e = blockIdx.x * 256 + threadIdx.x;
  float acc = bias2[e & (D_OUT - 1)];
#pragma unroll
  for (int s = 0; s < SPLITK; ++s)
    acc += zpart[(size_t)s * BATCH * D_OUT + e];
  zbf[e] = f32_to_bf16(fmaxf(acc, 0.f));
}

// ---------------------------------------------------------------------------
// Kernel 5: out = z_bf16 . wf2sT^T + bf2 (f32). Same async double-buffering.
// ---------------------------------------------------------------------------
__global__ __launch_bounds__(256) void gemm3_kernel(
    const unsigned short* __restrict__ zbf, const unsigned short* __restrict__ wBT,
    const float* __restrict__ bf2, float* __restrict__ out) {
  __shared__ unsigned short ldsA[2][128 * 32];
  __shared__ unsigned short ldsB[2][128 * 32];
  const int c0 = blockIdx.x * 128;
  const int r0 = blockIdx.y * 128;
  const int tid = threadIdx.x;
  const int lane = tid & 31, wave = tid >> 5;
  const int waveRow = wave & 3, waveCol = wave >> 2;
  const int laneIdx = lane & 15, laneHalf = lane >> 4;
  const int iters = D_OUT / 32;

  v8f acc[2][4] = {};

  auto stage = [&](int buf, int kk) {
#pragma unroll
    for (int i = 0; i < 2; ++i) {
      int idx = tid + i * 256;
      int row = idx >> 2, kc = idx & 3;
      copy16(zbf + (size_t)(r0 + row) * D_OUT + kk + kc * 8,
             &ldsA[buf][row * 32 + kc * 8]);
    }
#pragma unroll
    for (int i = 0; i < 2; ++i) {
      int idx = tid + i * 256;
      int row = idx >> 2, kc = idx & 3;
      copy16(wBT + (size_t)(c0 + row) * D_OUT + kk + kc * 8,
             &ldsB[buf][row * 32 + kc * 8]);
    }
  };

  stage(0, 0);
  for (int kt = 0; kt < iters; ++kt) {
    int buf = kt & 1;
    if (kt + 1 < iters) {
      stage(buf ^ 1, (kt + 1) * 32);
      async_wait(4);
    } else {
      async_wait(0);
    }
    __syncthreads();
    wmma_tile8(ldsA[buf], ldsB[buf], waveRow, waveCol, laneIdx, laneHalf, acc);
    __syncthreads();
  }

#pragma unroll
  for (int j = 0; j < 4; ++j) {
    int colo = c0 + waveCol * 64 + j * 16 + laneIdx;
    float bias_v = bf2[colo];
#pragma unroll
    for (int i = 0; i < 2; ++i) {
      int rowBase = r0 + waveRow * 32 + i * 16 + laneHalf * 8;
#pragma unroll
      for (int r = 0; r < 8; ++r)
        out[(size_t)(rowBase + r) * D_OUT + colo] = acc[i][j][r] + bias_v;
    }
  }
}

// ---------------------------------------------------------------------------
extern "C" void kernel_launch(void* const* d_in, const int* in_sizes, int n_in,
                              void* d_out, int out_size, void* d_ws, size_t ws_size,
                              hipStream_t stream) {
  const float* x     = (const float*)d_in[0];
  const float* W     = (const float*)d_in[1];
  const float* b     = (const float*)d_in[2];
  const float* gamma = (const float*)d_in[3];
  const float* beta  = (const float*)d_in[4];
  const float* Wf1   = (const float*)d_in[5];
  const float* bf1   = (const float*)d_in[6];
  const float* Wf2   = (const float*)d_in[7];
  const float* bf2   = (const float*)d_in[8];
  float* out = (float*)d_out;

  char* ws = (char*)d_ws;
  size_t off = 0;
  auto alloc = [&](size_t bytes) -> char* {
    char* p = ws + off;
    off += (bytes + 255) & ~(size_t)255;
    return p;
  };
  unsigned short* h_bf   = (unsigned short*)alloc((size_t)BATCH * KO * 2);     // 256 MB
  unsigned short* wf1sT  = (unsigned short*)alloc((size_t)KO * D_OUT * 2);     // 128 MB
  unsigned short* wf2sT  = (unsigned short*)alloc((size_t)D_OUT * D_OUT * 2);  //   2 MB
  unsigned short* z_bf   = (unsigned short*)alloc((size_t)BATCH * D_OUT * 2);  //   4 MB
  float* scale = (float*)alloc((size_t)KO * 4);
  float* shift = (float*)alloc((size_t)KO * 4);
  float* partB = (float*)alloc((size_t)64 * D_OUT * 4);
  float* bias2 = (float*)alloc((size_t)D_OUT * 4);
  float* zpart = (float*)alloc((size_t)SPLITK * BATCH * D_OUT * 4);            //  64 MB

  gemm1_kernel<<<dim3(D_OUT / 128, BATCH / 128, K_BRANCH), 256, 0, stream>>>(
      x, W, b, h_bf);
  stats_kernel<<<dim3(KO / 128), 256, 0, stream>>>(h_bf, gamma, beta, scale, shift);
  wf1prep_kernel<<<dim3(D_OUT / 256, 64), 256, 0, stream>>>(Wf1, scale, shift,
                                                            wf1sT, partB);
  bias2_reduce_kernel<<<dim3(D_OUT / 256), 256, 0, stream>>>(partB, bf1, bias2);
  wf2prep_kernel<<<dim3((D_OUT * D_OUT) / 256), 256, 0, stream>>>(Wf2, wf2sT);
  gemm2_kernel<<<dim3(D_OUT / 128, BATCH / 128, SPLITK), 256, 0, stream>>>(
      h_bf, wf1sT, zpart);
  zreduce_kernel<<<dim3((BATCH * D_OUT) / 256), 256, 0, stream>>>(zpart, bias2,
                                                                  z_bf);
  gemm3_kernel<<<dim3(D_OUT / 128, BATCH / 128), 256, 0, stream>>>(z_bf, wf2sT,
                                                                   bf2, out);
}
